// Head_22411139351074
// MI455X (gfx1250) — matrix operand order repaired and checked by reading, not compiled, not run on hardware
//
#include <hip/hip_runtime.h>
#include <hip/hip_bf16.h>
#include <math.h>

typedef _Float16 f16;
typedef __attribute__((ext_vector_type(16))) _Float16 v16h;
typedef __attribute__((ext_vector_type(8)))  _Float16 v8h;
typedef __attribute__((ext_vector_type(8)))  float    v8f;
typedef __attribute__((ext_vector_type(4)))  float    v4f;

#define WMMA_F16(a, b, c) \
    __builtin_amdgcn_wmma_f32_16x16x32_f16(false, (a), false, (b), (short)0, (c), false, false)

// ---------------------------------------------------------------------------
// Sizes
// ---------------------------------------------------------------------------
#define N_EMBD   1024
#define HEADSZ   64
#define SEQ      2048
#define NBATCH   8
#define MROWS    (NBATCH * SEQ)          // 16384

// Workspace layout (bytes)
#define WSWZ_ELEMS_PER_W  (32 * 4 * 32 * 16)             // kt * nt * lane * 16 f16 = 65536
#define OFF_WSWZ   0
#define OFF_Q16    (3 * WSWZ_ELEMS_PER_W * 2)            // 393216
#define OFF_K16    (OFF_Q16 + MROWS * HEADSZ * 2)        // +2MB
#define OFF_VT16   (OFF_K16 + MROWS * HEADSZ * 2)        // +2MB

// ---------------------------------------------------------------------------
// Kernel 1: swizzle Wk/Wq/Wv into WMMA B-operand layout (f16).
// B (32K x 16N) lane layout: group G=l>>4 holds K = {8G+0..7,
// 16+8G+0..7}. One contiguous 16-f16 run per lane.
// grid.x = 3 * 32 * 4  (w, ktile, ntile), block = 32
// ---------------------------------------------------------------------------
__global__ void swz_kernel(const float* __restrict__ Wk,
                           const float* __restrict__ Wq,
                           const float* __restrict__ Wv,
                           f16* __restrict__ wswzAll) {
    const int l  = threadIdx.x;
    const int G  = l >> 4;
    const int n  = l & 15;
    const int bid = blockIdx.x;
    const int w  = bid >> 7;
    const int kt = (bid >> 2) & 31;
    const int nt = bid & 3;
    const float* W = (w == 0) ? Wk : (w == 1) ? Wq : Wv;
    f16* dst = wswzAll + (size_t)w * WSWZ_ELEMS_PER_W
                       + ((size_t)(kt * 4 + nt) * 32 + l) * 16;
#pragma unroll
    for (int e = 0; e < 16; ++e) {
        const int kk = (e < 8) ? (8 * G + e) : (16 + 8 * G + (e - 8));
        dst[e] = (f16)W[(size_t)(kt * 32 + kk) * HEADSZ + nt * 16 + n];
    }
}

// ---------------------------------------------------------------------------
// Kernel 2: projection GEMM. One wave -> 16 rows x 64 cols of one of k/q/v.
// A (16M x 32K f16) layout: lane l holds row (l&15); group G=l>>4 holds
// K = {8G+0..7, 16+8G+0..7}  -> two contiguous 8-float runs of x.
// grid = (1024, 3), block = 32
// ---------------------------------------------------------------------------
__global__ void proj_kernel(const float* __restrict__ x,
                            const f16* __restrict__ wswzAll,
                            f16* __restrict__ k16,
                            f16* __restrict__ q16,
                            f16* __restrict__ vT16) {
    const int l = threadIdx.x;
    const int G = l >> 4;
    const int n = l & 15;
    const int w = blockIdx.y;                 // 0=k, 1=q, 2=v
    const int r0 = blockIdx.x * 16;

    const f16* wswz = wswzAll + (size_t)w * WSWZ_ELEMS_PER_W;
    const float* xrow = x + (size_t)(r0 + n) * N_EMBD;

    v8f acc[4] = {};

    for (int kt = 0; kt < 32; ++kt) {
        const int off0 = kt * 32 + 8 * G;
        v4f f0 = *(const v4f*)(xrow + off0);
        v4f f1 = *(const v4f*)(xrow + off0 + 4);
        v4f f2 = *(const v4f*)(xrow + off0 + 16);
        v4f f3 = *(const v4f*)(xrow + off0 + 20);
        v16h a;
#pragma unroll
        for (int e = 0; e < 4; ++e) {
            a[e]      = (f16)f0[e];
            a[e + 4]  = (f16)f1[e];
            a[e + 8]  = (f16)f2[e];
            a[e + 12] = (f16)f3[e];
        }
        const f16* wb = wswz + ((size_t)kt * 4 * 32 + l) * 16;
#pragma unroll
        for (int nt = 0; nt < 4; ++nt) {
            v16h b = *(const v16h*)(wb + (size_t)nt * 32 * 16);
            acc[nt] = WMMA_F16(a, b, acc[nt]);
        }
    }

    const float sc = (w == 1) ? 0.03125f : 1.0f;   // fold C^-0.5 into q

    if (w == 2) {
        // store v transposed: vT[b][h][t_local]
#pragma unroll
        for (int nt = 0; nt < 4; ++nt) {
#pragma unroll
            for (int j = 0; j < 8; ++j) {
                const int t  = r0 + j + 8 * G;
                const int b  = t >> 11;
                const int tl = t & (SEQ - 1);
                const int h  = nt * 16 + n;
                vT16[((size_t)b * HEADSZ + h) * SEQ + tl] = (f16)acc[nt][j];
            }
        }
    } else {
        f16* dst = (w == 0) ? k16 : q16;
#pragma unroll
        for (int nt = 0; nt < 4; ++nt) {
#pragma unroll
            for (int j = 0; j < 8; ++j) {
                dst[(size_t)(r0 + j + 8 * G) * HEADSZ + nt * 16 + n] =
                    (f16)(acc[nt][j] * sc);
            }
        }
    }
}

// ---------------------------------------------------------------------------
// A-operand loader from row-major f16 [.,64]: two contiguous 16B chunks.
// ---------------------------------------------------------------------------
__device__ inline v16h load_a16(const f16* base, int row, int kOff, int G) {
    v8h lo = *(const v8h*)(base + (size_t)row * HEADSZ + kOff + 16 * G);
    v8h hi = *(const v8h*)(base + (size_t)row * HEADSZ + kOff + 16 * G + 8);
    return __builtin_shufflevector(lo, hi, 0, 1, 2, 3, 4, 5, 6, 7,
                                           8, 9, 10, 11, 12, 13, 14, 15);
}

// ---------------------------------------------------------------------------
// Kernel 3: flash attention, one wave per (batch, 16-row t-tile).
// Computes S^T = k @ q^T so the score C-layout *is* the A-layout of P
// for the P @ v WMMA (no transpose needed). Online softmax over s chunks
// of 32, causal-masked, stops at the diagonal.
// grid = (128, 8), block = 32
// ---------------------------------------------------------------------------
__global__ void attn_kernel(const f16* __restrict__ q16,
                            const f16* __restrict__ k16,
                            const f16* __restrict__ vT16,
                            float* __restrict__ out) {
    const int l = threadIdx.x;
    const int G = l >> 4;
    const int n = l & 15;
    const int rt = blockIdx.x;
    const int b  = blockIdx.y;
    const int tBase = rt * 16;

    const f16* qb = q16 + (size_t)b * SEQ * HEADSZ;
    const f16* kb = k16 + (size_t)b * SEQ * HEADSZ;
    const f16* vb = vT16 + (size_t)b * HEADSZ * SEQ;

    // q as B-operand (kk = h, n = t): contiguous 16 f16 per lane. Hoisted.
    v16h bq0 = *(const v16h*)(qb + (size_t)(tBase + n) * HEADSZ + 16 * G);
    v16h bq1 = *(const v16h*)(qb + (size_t)(tBase + n) * HEADSZ + 32 + 16 * G);

    float m = -INFINITY;
    float lsum = 0.0f;
    v8f O[4] = {};
    const int t = tBase + n;               // this lane's softmax row
    const int nChunks = (rt + 2) >> 1;     // ceil((rt+1)/2)

    for (int c = 0; c < nChunks; ++c) {
        const int sBase = c * 32;

        // k as A-operand for two 16-wide s sub-tiles, two 32-wide h chunks
        v16h a00 = load_a16(kb, sBase + n,      0,  G);
        v16h a01 = load_a16(kb, sBase + n,      32, G);
        v16h a10 = load_a16(kb, sBase + 16 + n, 0,  G);
        v16h a11 = load_a16(kb, sBase + 16 + n, 32, G);

        v8f s0 = {}, s1 = {};
        s0 = WMMA_F16(a00, bq0, s0);
        s0 = WMMA_F16(a01, bq1, s0);
        s1 = WMMA_F16(a10, bq0, s1);
        s1 = WMMA_F16(a11, bq1, s1);

        // causal mask + chunk row-max (row = t, s across regs + lane^16)
        float cmax = -INFINITY;
#pragma unroll
        for (int j = 0; j < 8; ++j) {
            const int si0 = sBase + j + 8 * G;
            const int si1 = si0 + 16;
            const float v0 = (si0 <= t) ? s0[j] : -INFINITY;
            const float v1 = (si1 <= t) ? s1[j] : -INFINITY;
            s0[j] = v0;
            s1[j] = v1;
            cmax = fmaxf(cmax, fmaxf(v0, v1));
        }
        cmax = fmaxf(cmax, __shfl_xor(cmax, 16, 32));
        const float mnew = fmaxf(m, cmax);
        const float escale = __expf(m - mnew);   // first chunk: exp(-inf)=0
        m = mnew;

        // P = exp(S - m), packed directly into A-operand layout:
        // score reg j of sub-tile i  ->  v16h element j + 8*i
        v16h pa;
        float psum = 0.0f;
#pragma unroll
        for (int j = 0; j < 8; ++j) {
            const float p0 = __expf(s0[j] - mnew);
            const float p1 = __expf(s1[j] - mnew);
            pa[j]     = (f16)p0;
            pa[8 + j] = (f16)p1;
            psum += p0 + p1;
        }
        psum += __shfl_xor(psum, 16, 32);
        lsum = lsum * escale + psum;

        // rescale O rows (row index of O reg j is tBase + j + 8G -> lane j+8G)
#pragma unroll
        for (int j = 0; j < 8; ++j) {
            const float fj = __shfl(escale, j + 8 * G, 32);
            O[0][j] *= fj; O[1][j] *= fj; O[2][j] *= fj; O[3][j] *= fj;
        }

        // O += P @ v   (v^T as B-operand: two 8-f16 runs per lane, s = 8G+0..7 and 16+8G+0..7)
#pragma unroll
        for (int nt = 0; nt < 4; ++nt) {
            v8h bvlo = *(const v8h*)(vb + (size_t)(nt * 16 + n) * SEQ
                                        + sBase + 8 * G);
            v8h bvhi = *(const v8h*)(vb + (size_t)(nt * 16 + n) * SEQ
                                        + sBase + 16 + 8 * G);
            v16h bv = __builtin_shufflevector(bvlo, bvhi, 0, 1, 2, 3, 4, 5, 6, 7,
                                              8, 9, 10, 11, 12, 13, 14, 15);
            O[nt] = WMMA_F16(pa, bv, O[nt]);
        }
    }

    const float inv = 1.0f / lsum;
#pragma unroll
    for (int j = 0; j < 8; ++j) {
        const float fj = __shfl(inv, j + 8 * G, 32);
        const int trow = tBase + j + 8 * G;
#pragma unroll
        for (int nt = 0; nt < 4; ++nt) {
            out[((size_t)b * SEQ + trow) * HEADSZ + nt * 16 + n] = O[nt][j] * fj;
        }
    }
}

// ---------------------------------------------------------------------------
extern "C" void kernel_launch(void* const* d_in, const int* in_sizes, int n_in,
                              void* d_out, int out_size, void* d_ws, size_t ws_size,
                              hipStream_t stream) {
    const float* x  = (const float*)d_in[0];
    const float* Wk = (const float*)d_in[1];
    const float* Wq = (const float*)d_in[2];
    const float* Wv = (const float*)d_in[3];
    float* out = (float*)d_out;

    char* ws = (char*)d_ws;
    f16* wswz = (f16*)(ws + OFF_WSWZ);
    f16* q16  = (f16*)(ws + OFF_Q16);
    f16* k16  = (f16*)(ws + OFF_K16);
    f16* vT16 = (f16*)(ws + OFF_VT16);

    swz_kernel <<<dim3(3 * 32 * 4), 32, 0, stream>>>(Wk, Wq, Wv, wswz);
    proj_kernel<<<dim3(MROWS / 16, 3), 32, 0, stream>>>(x, wswz, k16, q16, vT16);
    attn_kernel<<<dim3(SEQ / 16, NBATCH), 32, 0, stream>>>(q16, k16, vT16, out);
}
